// WindowAttention4D_69758858821848
// MI455X (gfx1250) — compile-verified, matrix operand order checked by reading, not run on hardware
//
#include <hip/hip_runtime.h>
#include <math.h>

typedef unsigned int  u32;
typedef unsigned short u16;
typedef unsigned long long u64;

typedef __attribute__((ext_vector_type(16))) __bf16 bf16x16;
typedef __attribute__((ext_vector_type(8)))  __bf16 bf16x8;
typedef __attribute__((ext_vector_type(8)))  float  f32x8;

// ---------- bf16 helpers (manual, no header dependence) ----------
__device__ __forceinline__ float bf2f(u16 v) {
    union { u32 u; float f; } x; x.u = ((u32)v) << 16; return x.f;
}
__device__ __forceinline__ u16 f2bfu(float f) {
    union { float f; u32 u; } x; x.f = f;
    u32 u = x.u;
    u += 0x7fffu + ((u >> 16) & 1u);   // round-to-nearest-even
    return (u16)(u >> 16);
}
__device__ __forceinline__ __bf16 f2bf(float f) {
    union { u16 u; __bf16 b; } x; x.u = f2bfu(f); return x.b;
}

// Load a 16x32(K) bf16 WMMA fragment: caller passes pointer already offset by
// (row-or-col index)*stride + kbase + half*8 ; chunks at +0 and +16 elements.
__device__ __forceinline__ bf16x16 ldfrag(const __bf16* p) {
    union { bf16x16 v; bf16x8 h[2]; } u;
    u.h[0] = *(const bf16x8*)(p);
    u.h[1] = *(const bf16x8*)(p + 16);
    return u.v;
}

__device__ __forceinline__ f32x8 wmma_bf16(bf16x16 a, bf16x16 b, f32x8 c) {
    return __builtin_amdgcn_wmma_f32_16x16x32_bf16(false, a, false, b, (short)0, c, false, false);
}

// Async global->LDS DMA of one 16B chunk (CDNA5 GLOBAL_LOAD_ASYNC_TO_LDS_B128,
// GV addressing: vdst = LDS byte offset, vaddr = 64-bit global address).
// Tracked by ASYNCcnt; pair with async_wait() before the workgroup barrier.
__device__ __forceinline__ void async_b128(void* lds_dst, const void* gsrc) {
    u32 lds_off = (u32)(size_t)lds_dst;   // low 32 bits of flat addr = LDS offset
    asm volatile("global_load_async_to_lds_b128 %0, %1, off"
                 :: "v"(lds_off), "v"((u64)(size_t)gsrc) : "memory");
}
__device__ __forceinline__ void async_wait() {
    asm volatile("s_wait_asynccnt 0x0" ::: "memory");
}

// ---------- problem constants ----------
#define DIMC   128
#define NHEADS 4
#define HD     32
#define WA     256
#define NWIN   256     // 2*2*8*8 windows
#define SPATIAL 65536  // 4*4*64*64

// ---------------------------------------------------------------------------
// K1: weight conversion (f32 -> bf16), qkv bias vector, per-head logit scale
// ---------------------------------------------------------------------------
__global__ __launch_bounds__(256) void k_prep(
    const float* __restrict__ w_qkv, const float* __restrict__ proj_w,
    const float* __restrict__ q_bias, const float* __restrict__ v_bias,
    const float* __restrict__ logit_scale,
    __bf16* __restrict__ wq_bf, __bf16* __restrict__ wp_bf,
    float* __restrict__ qkvb, float* __restrict__ scale)
{
    int gid = blockIdx.x * 256 + threadIdx.x;
#pragma unroll
    for (int i = 0; i < 4; ++i) {
        int idx = i * 16384 + gid;             // 65536 total conversions
        if (idx < 49152) {
            wq_bf[idx] = f2bf(w_qkv[idx]);     // 384x128
        } else {
            int j = idx - 49152;               // 16384 proj weights
            wp_bf[j] = f2bf(proj_w[j]);
        }
    }
    if (blockIdx.x == 0) {
        int t = threadIdx.x;
        if (t < 128)      { qkvb[t] = q_bias[t]; qkvb[128 + t] = 0.0f; qkvb[256 + t] = v_bias[t]; }
        if (t < NHEADS)   { scale[t] = __expf(fminf(logit_scale[t], 4.6051702f)); } // log(100)
    }
}

// ---------------------------------------------------------------------------
// K2: CPB MLP table tab[2025][4]  (relu(ctab @ w1^T + b1) @ w2^T)
// ---------------------------------------------------------------------------
__device__ __forceinline__ float cpb_coord(int i, int w) {
    float c = (float)(i - (w - 1)) * 8.0f / ((float)(w - 1) + 1e-9f);
    float a = log2f(fabsf(c) + 1.0f) * (1.0f / 3.0f);   // /log2(8)
    return copysignf(a, c);
}
__global__ __launch_bounds__(256) void k_cpb(
    const float* __restrict__ w1, const float* __restrict__ b1,
    const float* __restrict__ w2, float* __restrict__ tab)
{
    int idx = blockIdx.x * 256 + threadIdx.x;
    if (idx >= 2025) return;
    int i0 = idx / 675, r = idx % 675;
    int i1 = r / 225;   r %= 225;
    int i2 = r / 15;
    int i3 = r % 15;
    float c0 = cpb_coord(i0, 2), c1 = cpb_coord(i1, 2);
    float c2 = cpb_coord(i2, 8), c3 = cpb_coord(i3, 8);
    float acc0 = 0.f, acc1 = 0.f, acc2 = 0.f, acc3 = 0.f;
    for (int j = 0; j < 512; ++j) {
        const float* wr = w1 + j * 4;
        float h = fmaxf(c0 * wr[0] + c1 * wr[1] + c2 * wr[2] + c3 * wr[3] + b1[j], 0.0f);
        acc0 += w2[0 * 512 + j] * h;
        acc1 += w2[1 * 512 + j] * h;
        acc2 += w2[2 * 512 + j] * h;
        acc3 += w2[3 * 512 + j] * h;
    }
    tab[idx * 4 + 0] = acc0; tab[idx * 4 + 1] = acc1;
    tab[idx * 4 + 2] = acc2; tab[idx * 4 + 3] = acc3;
}

// ---------------------------------------------------------------------------
// K3: rpb[h][i][j] = 16 * sigmoid(tab[relidx(i,j)][h])   (4 x 256 x 256 f32)
// ---------------------------------------------------------------------------
__global__ __launch_bounds__(256) void k_rpb(
    const float* __restrict__ tab, float* __restrict__ rpb)
{
    int i = blockIdx.x, j = threadIdx.x;
    int a0 = (i >> 7) & 1, a1 = (i >> 6) & 1, a2 = (i >> 3) & 7, a3 = i & 7;
    int b0 = (j >> 7) & 1, b1 = (j >> 6) & 1, b2 = (j >> 3) & 7, b3 = j & 7;
    int d0 = a0 - b0 + 1, d1 = a1 - b1 + 1, d2 = a2 - b2 + 7, d3 = a3 - b3 + 7;
    int ridx = d0 * 675 + d1 * 225 + d2 * 15 + d3;
#pragma unroll
    for (int h = 0; h < NHEADS; ++h) {
        float v = tab[ridx * 4 + h];
        rpb[((size_t)h << 16) + ((size_t)i << 8) + j] = 16.0f / (1.0f + __expf(-v));
    }
}

// ---------------------------------------------------------------------------
// K4: shifted-window gather + QKV GEMM (bf16 WMMA, K=128), write q/k/v bf16
//     grid: 512 blocks (window * half-window of 128 tokens), 256 threads
// ---------------------------------------------------------------------------
__global__ __launch_bounds__(256) void k_qkv(
    const float* __restrict__ x, const __bf16* __restrict__ wq_bf,
    const float* __restrict__ qkvb,
    __bf16* __restrict__ qg, __bf16* __restrict__ kg, __bf16* __restrict__ vg)
{
    __shared__ __align__(16) __bf16 xw[128 * 136];  // 128 tokens x 128 ch, pad stride 136

    const int tid = threadIdx.x;
    const int b   = blockIdx.x >> 1;
    const int hw  = blockIdx.x & 1;

    // window coords
    const int nw = b & 7, nh = (b >> 3) & 7, nl = (b >> 6) & 1, nt = (b >> 7) & 1;

    // gather: token tl = tid&127, channels looped
    const int tl = tid & 127;
    const int t  = hw * 128 + tl;
    const int t0 = (t >> 7) & 1, t1 = (t >> 6) & 1, t2 = (t >> 3) & 7, t3 = t & 7;
    const int T  = (nt * 2 + t0 + 1) & 3;
    const int L  = (nl * 2 + t1 + 1) & 3;
    const int Hc = (nh * 8 + t2 + 4) & 63;
    const int Wc = (nw * 8 + t3 + 4) & 63;
    const int sidx = T * 16384 + L * 4096 + Hc * 64 + Wc;

    for (int it = 0; it < 64; ++it) {
        int c = it * 2 + (tid >> 7);
        float v = x[(size_t)c * SPATIAL + sidx];
        xw[tl * 136 + c] = f2bf(v);
    }
    __syncthreads();

    const int wave = tid >> 5, lane = tid & 31;
    const int half = lane >> 4, l16 = lane & 15;

    for (int i = 0; i < 24; ++i) {
        int tt = i * 8 + wave;
        int tm = tt & 7;        // 8 row tiles (128 tokens)
        int tn = tt >> 3;       // 24 col tiles (384 outputs)
        f32x8 acc = {};
#pragma unroll
        for (int kk = 0; kk < 4; ++kk) {
            int kb = kk * 32;
            bf16x16 aF = ldfrag(&xw[(tm * 16 + l16) * 136 + kb + half * 8]);
            bf16x16 bF = ldfrag(wq_bf + (size_t)(tn * 16 + l16) * 128 + kb + half * 8);
            acc = wmma_bf16(aF, bF, acc);
        }
        int col   = tn * 16 + l16;
        int which = col >> 7;
        int cc    = col & 127;
        int head  = cc >> 5;
        int hd    = cc & 31;
        float bias = qkvb[col];
        __bf16* dst = (which == 0) ? qg : ((which == 1) ? kg : vg);
#pragma unroll
        for (int r = 0; r < 8; ++r) {
            int m = hw * 128 + tm * 16 + r + 8 * half;
            dst[((size_t)(b * NHEADS + head) * WA + m) * HD + hd] = f2bf(acc[r] + bias);
        }
    }
}

// ---------------------------------------------------------------------------
// K5: cosine-normalize q,k per (b,h,token); fold logit scale into q
// ---------------------------------------------------------------------------
__global__ __launch_bounds__(256) void k_norm(
    __bf16* __restrict__ qg, __bf16* __restrict__ kg,
    const float* __restrict__ scale)
{
    int v  = blockIdx.x * 256 + threadIdx.x;   // 256*4*256 vectors
    int h  = (v >> 8) & 3;
    u32* qp = (u32*)(qg) + (size_t)v * 16;     // 32 bf16 = 16 dwords
    u32* kp = (u32*)(kg) + (size_t)v * 16;
    float sc = scale[h];

    float qv[32], kv[32];
    float qs = 0.f, ks = 0.f;
#pragma unroll
    for (int i = 0; i < 16; ++i) {
        u32 qa = qp[i], ka = kp[i];
        float q0 = bf2f((u16)qa), q1 = bf2f((u16)(qa >> 16));
        float k0 = bf2f((u16)ka), k1 = bf2f((u16)(ka >> 16));
        qv[2*i] = q0; qv[2*i+1] = q1; kv[2*i] = k0; kv[2*i+1] = k1;
        qs += q0*q0 + q1*q1; ks += k0*k0 + k1*k1;
    }
    float qr = sc / fmaxf(sqrtf(qs), 1e-12f);
    float kr = 1.0f / fmaxf(sqrtf(ks), 1e-12f);
#pragma unroll
    for (int i = 0; i < 16; ++i) {
        u32 qa = ((u32)f2bfu(qv[2*i+1] * qr) << 16) | f2bfu(qv[2*i] * qr);
        u32 ka = ((u32)f2bfu(kv[2*i+1] * kr) << 16) | f2bfu(kv[2*i] * kr);
        qp[i] = qa; kp[i] = ka;
    }
}

// ---------------------------------------------------------------------------
// K6: flash attention per (window, head). 8 waves, each owns 2 row tiles,
//     streams 8 key blocks of 32 tokens with online softmax.
//     K staged via async global->LDS DMA (ASYNCcnt); V staged transposed.
// ---------------------------------------------------------------------------
__global__ __launch_bounds__(256) void k_attn(
    const __bf16* __restrict__ qg, const __bf16* __restrict__ kg,
    const __bf16* __restrict__ vg, const float* __restrict__ rpb,
    __bf16* __restrict__ ao)
{
    __shared__ __align__(16) __bf16 kL [256 * 40];      // K: token-major, stride 40
    __shared__ __align__(16) __bf16 vT [32 * 264];      // V^T: hd-major, stride 264
    __shared__ __align__(16) __bf16 Pst[8 * 16 * 40];   // per-wave P staging 16x32
    __shared__ int lab[256];

    const int tid = threadIdx.x;
    const int b = blockIdx.x >> 2;
    const int h = blockIdx.x & 3;
    const size_t bh = (size_t)(b * NHEADS + h) * WA;

    // stage K via async DMA: 256 rows of 64B -> LDS stride 80B, 16B chunks.
    {
        const char* src = (const char*)(kg + bh * HD);
#pragma unroll
        for (int i = 0; i < 4; ++i) {
            int idx = i * 256 + tid;          // 1024 chunks
            int row = idx >> 2, c = idx & 3;
            async_b128((char*)kL + row * 80 + c * 16, src + row * 64 + c * 16);
        }
    }
    // stage V transposed (format change -> VGPR path)
    {
        const __bf16* src = vg + bh * HD;
        for (int i = 0; i < 32; ++i) {
            int idx = i * 256 + tid;
            int tt = idx >> 5, d = idx & 31;
            vT[d * 264 + tt] = src[idx];
        }
    }
    // region labels for shifted-window mask
    {
        int t = tid;
        int nw = b & 7, nh_ = (b >> 3) & 7, nl = (b >> 6) & 1, nt = (b >> 7) & 1;
        int t0 = (t >> 7) & 1, t1 = (t >> 6) & 1, t2 = (t >> 3) & 7, t3 = t & 7;
        int r0 = (nt == 1) ? ((t0 < 1) ? 1 : 2) : 0;
        int r1 = (nl == 1) ? ((t1 < 1) ? 1 : 2) : 0;
        int r2 = (nh_ == 7) ? ((t2 < 4) ? 1 : 2) : 0;
        int r3 = (nw == 7) ? ((t3 < 4) ? 1 : 2) : 0;
        lab[t] = ((r0 * 3 + r1) * 3 + r2) * 3 + r3;
    }
    async_wait();          // this wave's async LDS writes committed
    __syncthreads();       // publish to all waves

    const int wave = tid >> 5, lane = tid & 31;
    const int half = lane >> 4, l16 = lane & 15;
    __bf16* myP = Pst + wave * 16 * 40;

    for (int mi = 0; mi < 2; ++mi) {
        const int mbase = (wave * 2 + mi) * 16;
        // q fragment straight from global (row stride 32 bf16 = 64B)
        bf16x16 qF = ldfrag(qg + (bh + mbase + l16) * HD + half * 8);

        float m_run[8], l_run[8];
        f32x8 accO0 = {}, accO1 = {};
#pragma unroll
        for (int r = 0; r < 8; ++r) { m_run[r] = -3.0e38f; l_run[r] = 0.0f; }

        int mrow[8];
#pragma unroll
        for (int r = 0; r < 8; ++r) mrow[r] = mbase + r + 8 * half;

        for (int nt2 = 0; nt2 < 8; ++nt2) {
            const int kb = nt2 * 32;
            // S sub-tiles (16x16 each), K = head dim 32 -> 1 wmma each
            bf16x16 kF0 = ldfrag(&kL[(kb + l16) * 40 + half * 8]);
            bf16x16 kF1 = ldfrag(&kL[(kb + 16 + l16) * 40 + half * 8]);
            f32x8 z = {};
            f32x8 s0 = wmma_bf16(qF, kF0, z);
            f32x8 s1 = wmma_bf16(qF, kF1, z);

            const int n0 = kb + l16, n1 = kb + 16 + l16;
            const int ln0 = lab[n0], ln1 = lab[n1];
#pragma unroll
            for (int r = 0; r < 8; ++r) {
                const float* rb = rpb + (((size_t)h << 16) + ((size_t)mrow[r] << 8));
                int lm = lab[mrow[r]];
                float sv0 = s0[r] + rb[n0] + ((lm != ln0) ? -100.0f : 0.0f);
                float sv1 = s1[r] + rb[n1] + ((lm != ln1) ? -100.0f : 0.0f);
                // row max over 32 cols (lanes of same half hold distinct cols)
                float tm = fmaxf(sv0, sv1);
                tm = fmaxf(tm, __shfl_xor(tm, 1, 32));
                tm = fmaxf(tm, __shfl_xor(tm, 2, 32));
                tm = fmaxf(tm, __shfl_xor(tm, 4, 32));
                tm = fmaxf(tm, __shfl_xor(tm, 8, 32));
                float nm = fmaxf(m_run[r], tm);
                float sc = __expf(m_run[r] - nm);
                m_run[r] = nm;
                float p0 = __expf(sv0 - nm);
                float p1 = __expf(sv1 - nm);
                float ts = p0 + p1;
                ts += __shfl_xor(ts, 1, 32);
                ts += __shfl_xor(ts, 2, 32);
                ts += __shfl_xor(ts, 4, 32);
                ts += __shfl_xor(ts, 8, 32);
                l_run[r] = l_run[r] * sc + ts;
                accO0[r] *= sc;
                accO1[r] *= sc;
                // stage P tile (D layout -> row-major LDS)
                __bf16* prow = myP + (r + 8 * half) * 40;
                prow[l16]      = f2bf(p0);
                prow[16 + l16] = f2bf(p1);
            }
            // O += P(16x32) @ V(32xHD): A from staging, B from vT
            bf16x16 pF = ldfrag(myP + l16 * 40 + half * 8);
            bf16x16 vF0 = ldfrag(&vT[(0  + l16) * 264 + kb + half * 8]);
            bf16x16 vF1 = ldfrag(&vT[(16 + l16) * 264 + kb + half * 8]);
            accO0 = wmma_bf16(pF, vF0, accO0);
            accO1 = wmma_bf16(pF, vF1, accO1);
        }
        // normalize & write (B_, WA, DIM) bf16 with col = h*32 + hd
#pragma unroll
        for (int r = 0; r < 8; ++r) {
            float rinv = 1.0f / l_run[r];
            size_t rowo = ((size_t)b * WA + mrow[r]) * DIMC + h * HD;
            ao[rowo + 0  + l16] = f2bf(accO0[r] * rinv);
            ao[rowo + 16 + l16] = f2bf(accO1[r] * rinv);
        }
    }
}

// ---------------------------------------------------------------------------
// K7: output projection GEMM + inverse window partition + un-shift scatter
//     grid: 512 blocks (window * half-window), 256 threads
//     A staged via async global->LDS DMA.
// ---------------------------------------------------------------------------
__global__ __launch_bounds__(256) void k_proj(
    const __bf16* __restrict__ ao, const __bf16* __restrict__ wp_bf,
    const float* __restrict__ proj_b, float* __restrict__ out)
{
    __shared__ __align__(16) __bf16 aL[128 * 136];

    const int tid = threadIdx.x;
    const int b  = blockIdx.x >> 1;
    const int hw = blockIdx.x & 1;
    const int nw = b & 7, nh = (b >> 3) & 7, nl = (b >> 6) & 1, nt = (b >> 7) & 1;

    // stage A via async DMA: 128 rows of 256B -> LDS stride 272B, 16B chunks.
    {
        const char* src = (const char*)(ao + ((size_t)b * WA + hw * 128) * DIMC);
#pragma unroll
        for (int i = 0; i < 8; ++i) {
            int idx = i * 256 + tid;          // 2048 chunks
            int row = idx >> 4, c = idx & 15;
            async_b128((char*)aL + row * 272 + c * 16, src + row * 256 + c * 16);
        }
    }
    async_wait();
    __syncthreads();

    const int wave = tid >> 5, lane = tid & 31;
    const int half = lane >> 4, l16 = lane & 15;

    for (int i = 0; i < 8; ++i) {
        int tt = i * 8 + wave;
        int tm = tt & 7, tn = tt >> 3;
        f32x8 acc = {};
#pragma unroll
        for (int kk = 0; kk < 4; ++kk) {
            int kb = kk * 32;
            bf16x16 aF = ldfrag(&aL[(tm * 16 + l16) * 136 + kb + half * 8]);
            bf16x16 bF = ldfrag(wp_bf + (size_t)(tn * 16 + l16) * 128 + kb + half * 8);
            acc = wmma_bf16(aF, bF, acc);
        }
        int col = tn * 16 + l16;
        float pb = proj_b[col];
#pragma unroll
        for (int r = 0; r < 8; ++r) {
            int t  = hw * 128 + tm * 16 + r + 8 * half;
            int t0 = (t >> 7) & 1, t1 = (t >> 6) & 1, t2 = (t >> 3) & 7, t3 = t & 7;
            int T  = (nt * 2 + t0 + 1) & 3;
            int L  = (nl * 2 + t1 + 1) & 3;
            int Hc = (nh * 8 + t2 + 4) & 63;
            int Wc = (nw * 8 + t3 + 4) & 63;
            out[(size_t)col * SPATIAL + T * 16384 + L * 4096 + Hc * 64 + Wc] = acc[r] + pb;
        }
    }
}

// ---------------------------------------------------------------------------
extern "C" void kernel_launch(void* const* d_in, const int* in_sizes, int n_in,
                              void* d_out, int out_size, void* d_ws, size_t ws_size,
                              hipStream_t stream) {
    const float* x      = (const float*)d_in[0];
    const float* w_qkv  = (const float*)d_in[1];
    const float* q_bias = (const float*)d_in[2];
    const float* v_bias = (const float*)d_in[3];
    const float* lscale = (const float*)d_in[4];
    const float* cpb_w1 = (const float*)d_in[5];
    const float* cpb_b1 = (const float*)d_in[6];
    const float* cpb_w2 = (const float*)d_in[7];
    const float* proj_w = (const float*)d_in[8];
    const float* proj_b = (const float*)d_in[9];
    float* out = (float*)d_out;

    char* ws = (char*)d_ws;
    size_t OFF_WQ    = 0;                      // 49152 bf16
    size_t OFF_WP    = OFF_WQ + 98304;         // 16384 bf16
    size_t OFF_QKVB  = OFF_WP + 32768;         // 384 f32
    size_t OFF_SCALE = OFF_QKVB + 1536;        // 4 f32
    size_t OFF_TAB   = (OFF_SCALE + 16 + 255) & ~(size_t)255;   // 2025*4 f32
    size_t OFF_RPB   = (OFF_TAB + 32400 + 255) & ~(size_t)255;  // 4*256*256 f32
    size_t OFF_Q     = (OFF_RPB + 1048576 + 255) & ~(size_t)255;
    size_t OFF_K     = OFF_Q + 16777216;
    size_t OFF_V     = OFF_K + 16777216;
    size_t OFF_AO    = OFF_V + 16777216;

    __bf16* wq_bf = (__bf16*)(ws + OFF_WQ);
    __bf16* wp_bf = (__bf16*)(ws + OFF_WP);
    float*  qkvb  = (float*)(ws + OFF_QKVB);
    float*  scale = (float*)(ws + OFF_SCALE);
    float*  tab   = (float*)(ws + OFF_TAB);
    float*  rpb   = (float*)(ws + OFF_RPB);
    __bf16* qg    = (__bf16*)(ws + OFF_Q);
    __bf16* kg    = (__bf16*)(ws + OFF_K);
    __bf16* vg    = (__bf16*)(ws + OFF_V);
    __bf16* ao    = (__bf16*)(ws + OFF_AO);

    k_prep<<<64, 256, 0, stream>>>(w_qkv, proj_w, q_bias, v_bias, lscale,
                                   wq_bf, wp_bf, qkvb, scale);
    k_cpb <<<8, 256, 0, stream>>>(cpb_w1, cpb_b1, cpb_w2, tab);
    k_rpb <<<256, 256, 0, stream>>>(tab, rpb);
    k_qkv <<<512, 256, 0, stream>>>(x, wq_bf, qkvb, qg, kg, vg);
    k_norm<<<1024, 256, 0, stream>>>(qg, kg, scale);
    k_attn<<<1024, 256, 0, stream>>>(qg, kg, vg, rpb, ao);
    k_proj<<<512, 256, 0, stream>>>(ao, wp_bf, proj_b, out);
}
